// CTCAHybrid_23218593202341
// MI455X (gfx1250) — matrix operand order repaired and checked by reading, not compiled, not run on hardware
//
#include <hip/hip_runtime.h>
#include <math.h>

// Problem constants
#define B_    64
#define L_    14
#define E_    768
#define S_    576
#define H_    12
#define HD_   64
#define F_    3072
#define MPAD  176          // H_*L_ = 168 padded to 11*16
#define MT_Q  11           // MPAD/16
#define NT_S  36           // S_/16
#define NT_E  48           // E_/16
#define TROWS 896          // B_*L_
#define MT_T  56           // TROWS/16
#define NT_F  192          // F_/16

typedef __attribute__((ext_vector_type(16))) __bf16 v16bf;
typedef __attribute__((ext_vector_type(8)))  __bf16 v8bf;
typedef __attribute__((ext_vector_type(8)))  float  v8f;

__device__ __forceinline__ __bf16 f2bf(float f) {
  unsigned u = __builtin_bit_cast(unsigned, f);
  unsigned r = u + 0x7FFFu + ((u >> 16) & 1u);   // round-to-nearest-even
  unsigned short s = (unsigned short)(r >> 16);
  return __builtin_bit_cast(__bf16, s);
}

__device__ __forceinline__ v8f wmma_bf16(v16bf a, v16bf b, v8f c) {
  return __builtin_amdgcn_wmma_f32_16x16x32_bf16(false, a, false, b, (short)0, c,
                                                 false, false);
}

// ---- WMMA fragment loaders (ISA 7.12.2 layouts, wave32) --------------------
// A (16x32, M x K) from row-major bf16 [*, lda]:
//   lane<16 : row m0+lane,    K = {k0..k0+7, k0+16..k0+23}
//   lane>=16: row m0+lane-16, K = {k0+8..k0+15, k0+24..k0+31}
__device__ __forceinline__ v16bf frag_a(const __bf16* A, int lda, int m0, int k0,
                                        int lane) {
  int ln = lane & 15, half = lane >> 4;
  const __bf16* p = A + (size_t)(m0 + ln) * lda + k0 + half * 8;
  v8bf lo = *reinterpret_cast<const v8bf*>(p);
  v8bf hi = *reinterpret_cast<const v8bf*>(p + 16);
  v16bf r;
#pragma unroll
  for (int i = 0; i < 8; ++i) { r[i] = lo[i]; r[8 + i] = hi[i]; }
  return r;
}

// B (32x16, K x N) where the source is row-major B^T (N x K): per lane
//   K = {k0+8h..k0+8h+7, k0+16+8h..k0+16+8h+7}, h = lane>=16
__device__ __forceinline__ v16bf frag_bt(const __bf16* Bt, int ldb, int n0, int k0,
                                         int lane) {
  int ln = lane & 15, half = lane >> 4;
  const __bf16* p = Bt + (size_t)(n0 + ln) * ldb + k0 + half * 8;
  v8bf lo = *reinterpret_cast<const v8bf*>(p);
  v8bf hi = *reinterpret_cast<const v8bf*>(p + 16);
  v16bf r;
#pragma unroll
  for (int i = 0; i < 8; ++i) { r[i] = lo[i]; r[8 + i] = hi[i]; }
  return r;
}

// ---- small helper kernels --------------------------------------------------
__global__ void k_f2bf(const float* in, __bf16* out, int n) {
  int i = blockIdx.x * 256 + threadIdx.x;
  if (i < n) out[i] = f2bf(in[i]);
}

// Tiled transpose: x fp32 [b][e][s]  ->  xbT bf16 [b][s][e]
// grid = B_ * (E_/64) * (S_/64), 256 threads, 64x64 tile (LDS padded 64x65)
__global__ void __launch_bounds__(256)
k_transpose(const float* x, __bf16* xbT) {
  int blk = blockIdx.x;
  int sb  = blk % (S_ / 64);
  int r   = blk / (S_ / 64);
  int eb  = r % (E_ / 64);
  int b   = r / (E_ / 64);
  __shared__ __bf16 tile[64][65];
  int tc = threadIdx.x & 63;       // fast index (s on read, e on write)
  int tr = threadIdx.x >> 6;       // 0..3
  const float* src = x + ((size_t)b * E_ + eb * 64) * S_ + sb * 64;
#pragma unroll
  for (int i = 0; i < 16; ++i) {
    int row = tr + 4 * i;          // e-offset
    tile[row][tc] = f2bf(src[(size_t)row * S_ + tc]);
  }
  __syncthreads();
  __bf16* dst = xbT + ((size_t)b * S_ + sb * 64) * E_ + eb * 64;
#pragma unroll
  for (int i = 0; i < 16; ++i) {
    int row = tr + 4 * i;          // s-offset
    dst[(size_t)row * E_ + tc] = tile[tc][row];
  }
}

// out[r,c] = bias[c] + dot(A[r,:K], W[c,:K])
__global__ void k_rowdot(const float* A, const float* W, const float* bias,
                         float* out, int rows, int cols, int K) {
  int i = blockIdx.x * 256 + threadIdx.x;
  if (i >= rows * cols) return;
  int r = i / cols, c = i % cols;
  const float* a = A + (size_t)r * K;
  const float* w = W + (size_t)c * K;
  float acc = bias ? bias[c] : 0.f;
#pragma unroll 4
  for (int k = 0; k < K; ++k) acc += a[k] * w[k];
  out[i] = acc;
}

// qkA[h*14+l, e] = 0.125 * sum_d q[l, h*64+d] * wk[h*64+d, e]   (rows>=168 -> 0)
__global__ void k_qkfold(const float* q, const float* wqkv, __bf16* qkA) {
  int i = blockIdx.x * 256 + threadIdx.x;          // MPAD*E_
  int row = i / E_, e = i % E_;
  float acc = 0.f;
  if (row < H_ * L_) {
    int h = row / L_, l = row % L_;
    const float* qq = q + (size_t)l * E_ + h * HD_;
    const float* wk = wqkv + (size_t)(E_ + h * HD_) * E_ + e;
#pragma unroll 4
    for (int d = 0; d < HD_; ++d) acc += qq[d] * wk[(size_t)d * E_];
    acc *= 0.125f;                                  // 1/sqrt(64)
  }
  qkA[i] = f2bf(acc);
}

// cls<->cls attention core (batch independent). blocks = H_*L_, 64 threads.
__global__ void k_ccattn(const float* qkv, float* cc_ctx) {
  int h = blockIdx.x / L_;
  int l = blockIdx.x % L_;
  int t = threadIdx.x;                              // 0..63 == head dim
  __shared__ float sc[L_];
  if (t < L_) {
    float acc = 0.f;
    const float* q = qkv + (size_t)l * (3 * E_) + h * HD_;
    const float* k = qkv + (size_t)t * (3 * E_) + E_ + h * HD_;
#pragma unroll 4
    for (int d = 0; d < HD_; ++d) acc += q[d] * k[d];
    sc[t] = acc * 0.125f;
  }
  __syncthreads();
  float mx = -1e30f;
  for (int s = 0; s < L_; ++s) mx = fmaxf(mx, sc[s]);
  float pr[L_]; float sum = 0.f;
  for (int s = 0; s < L_; ++s) { pr[s] = expf(sc[s] - mx); sum += pr[s]; }
  float inv = 1.f / sum;
  float acc = 0.f;
  for (int s = 0; s < L_; ++s)
    acc += pr[s] * qkv[(size_t)s * (3 * E_) + 2 * E_ + h * HD_ + t];
  cc_ctx[(size_t)l * E_ + h * HD_ + t] = acc * inv;
}

// ---- WMMA GEMM 1: scores[b] = qkA(176x768) @ xbT[b]^T  (16x32 strip/wave) --
__global__ void __launch_bounds__(256)
k_scores(const __bf16* qkA, const __bf16* xbT, float* scoresP) {
  int wave = blockIdx.x * 8 + (threadIdx.x >> 5);
  int lane = threadIdx.x & 31;
  const int NP = NT_S / 2;                          // 18 strips of 32
  int b  = wave / (MT_Q * NP);
  int r  = wave % (MT_Q * NP);
  int m0 = (r / NP) * 16, n0 = (r % NP) * 32;
  const __bf16* Xt = xbT + (size_t)b * S_ * E_;     // rows: s, cols: e
  v8f c0 = {0.f, 0.f, 0.f, 0.f, 0.f, 0.f, 0.f, 0.f};
  v8f c1 = c0;
  for (int k0 = 0; k0 < E_; k0 += 32) {
    v16bf a  = frag_a(qkA, E_, m0, k0, lane);
    v16bf b0 = frag_bt(Xt, E_, n0, k0, lane);
    v16bf b1 = frag_bt(Xt, E_, n0 + 16, k0, lane);
    c0 = wmma_bf16(a, b0, c0);
    c1 = wmma_bf16(a, b1, c1);
  }
  int ln = lane & 15, half = lane >> 4;
  float* out = scoresP + ((size_t)b * MPAD + m0 + half * 8) * S_ + n0 + ln;
#pragma unroll
  for (int v = 0; v < 8; ++v) {
    out[(size_t)v * S_] = c0[v];
    out[(size_t)v * S_ + 16] = c1[v];
  }
}

// softmax over S per (b,row); writes normalized fp32 back + bf16 A-matrix copy
__global__ void __launch_bounds__(192)
k_softmax(float* scoresP, __bf16* attnB) {
  int row = blockIdx.x % MPAD;
  int b   = blockIdx.x / MPAD;
  size_t base = ((size_t)b * MPAD + row) * S_;
  int t = threadIdx.x;                              // 192 threads * 3 = 576
  if (row >= H_ * L_) {                             // padded rows -> zeros
    attnB[base + t] = f2bf(0.f);
    attnB[base + t + 192] = f2bf(0.f);
    attnB[base + t + 384] = f2bf(0.f);
    return;
  }
  __shared__ float red[192];
  float v0 = scoresP[base + t];
  float v1 = scoresP[base + t + 192];
  float v2 = scoresP[base + t + 384];
  red[t] = fmaxf(v0, fmaxf(v1, v2));
  __syncthreads();
  for (int off = 96; off > 0; off >>= 1) {
    if (t < off) red[t] = fmaxf(red[t], red[t + off]);
    __syncthreads();
  }
  float mx = red[0];
  __syncthreads();
  float e0 = expf(v0 - mx), e1 = expf(v1 - mx), e2 = expf(v2 - mx);
  red[t] = e0 + e1 + e2;
  __syncthreads();
  for (int off = 96; off > 0; off >>= 1) {
    if (t < off) red[t] += red[t + off];
    __syncthreads();
  }
  float inv = 1.f / red[0];
  e0 *= inv; e1 *= inv; e2 *= inv;
  scoresP[base + t] = e0;       attnB[base + t] = f2bf(e0);
  scoresP[base + t + 192] = e1; attnB[base + t + 192] = f2bf(e1);
  scoresP[base + t + 384] = e2; attnB[base + t + 384] = f2bf(e2);
}

// out_attn[b,l,s] = mean_h attn[b,h,l,s]
__global__ void k_attnmean(const float* scoresP, float* out2) {
  int i = blockIdx.x * 256 + threadIdx.x;           // B_*L_*S_
  int s = i % S_; int r = i / S_; int l = r % L_; int b = r / L_;
  float acc = 0.f;
#pragma unroll
  for (int h = 0; h < H_; ++h)
    acc += scoresP[((size_t)b * MPAD + h * L_ + l) * S_ + s];
  out2[i] = acc * (1.f / (float)H_);
}

// ---- WMMA GEMM 2: ctx[b] = attn[b](176x576) @ xb[b]^T(576x768) -------------
__global__ void __launch_bounds__(256)
k_ctx(const __bf16* attnB, const __bf16* xb, float* ctxP) {
  int wave = blockIdx.x * 8 + (threadIdx.x >> 5);
  int lane = threadIdx.x & 31;
  const int NP = NT_E / 2;                          // 24 strips of 32
  int b  = wave / (MT_Q * NP);
  int r  = wave % (MT_Q * NP);
  int m0 = (r / NP) * 16, n0 = (r % NP) * 32;
  const __bf16* A = attnB + (size_t)b * MPAD * S_;
  const __bf16* X = xb + (size_t)b * E_ * S_;       // B^T rows: n = e, k = s
  v8f c0 = {0.f, 0.f, 0.f, 0.f, 0.f, 0.f, 0.f, 0.f};
  v8f c1 = c0;
  for (int k0 = 0; k0 < S_; k0 += 32) {
    v16bf a  = frag_a(A, S_, m0, k0, lane);
    v16bf b0 = frag_bt(X, S_, n0, k0, lane);
    v16bf b1 = frag_bt(X, S_, n0 + 16, k0, lane);
    c0 = wmma_bf16(a, b0, c0);
    c1 = wmma_bf16(a, b1, c1);
  }
  int ln = lane & 15, half = lane >> 4;
  float* out = ctxP + ((size_t)b * MPAD + m0 + half * 8) * E_ + n0 + ln;
#pragma unroll
  for (int v = 0; v < 8; ++v) {
    out[(size_t)v * E_] = c0[v];
    out[(size_t)v * E_ + 16] = c1[v];
  }
}

// ctxv[b,l,f] = bv[f] + sum_e ctx[b, h(f), l, e] * wv[f, e]
__global__ void k_vproj(const float* ctxP, const float* wqkv, const float* bqkv,
                        float* ctxv) {
  int i = blockIdx.x * 256 + threadIdx.x;           // B_*L_*E_
  int e = i % E_; int bl = i / E_; int l = bl % L_; int b = bl / L_;
  int h = e >> 6;
  const float* a = ctxP + ((size_t)b * MPAD + h * L_ + l) * E_;
  const float* w = wqkv + (size_t)(2 * E_ + e) * E_;
  float acc = bqkv[2 * E_ + e];
#pragma unroll 4
  for (int k = 0; k < E_; ++k) acc += a[k] * w[k];
  ctxv[i] = acc;
}

// tok0 = cls + (ctxv @ wo^T + bo) + cc_out
__global__ void k_oproj(const float* ctxv, const float* wo, const float* bo,
                        const float* cls, const float* ccout, float* tok0) {
  int i = blockIdx.x * 256 + threadIdx.x;           // B_*L_*E_
  int e = i % E_; int bl = i / E_; int l = bl % L_;
  const float* a = ctxv + (size_t)bl * E_;
  const float* w = wo + (size_t)e * E_;
  float acc = bo[e];
#pragma unroll 4
  for (int k = 0; k < E_; ++k) acc += a[k] * w[k];
  tok0[i] = acc + cls[(size_t)l * E_ + e] + ccout[(size_t)l * E_ + e];
}

// layernorm -> bf16 A-matrix for the MLP GEMMs
__global__ void __launch_bounds__(256)
k_layernorm(const float* tok0, const float* g, const float* bb, __bf16* hb) {
  int row = blockIdx.x;                             // 0..895
  int t = threadIdx.x;                              // 256 * 3 = 768
  const float* x = tok0 + (size_t)row * E_;
  float a0 = x[t], a1 = x[t + 256], a2 = x[t + 512];
  __shared__ float red[256];
  red[t] = a0 + a1 + a2;
  __syncthreads();
  for (int off = 128; off > 0; off >>= 1) {
    if (t < off) red[t] += red[t + off];
    __syncthreads();
  }
  float mu = red[0] * (1.f / (float)E_);
  __syncthreads();
  float d0 = a0 - mu, d1 = a1 - mu, d2 = a2 - mu;
  red[t] = d0 * d0 + d1 * d1 + d2 * d2;
  __syncthreads();
  for (int off = 128; off > 0; off >>= 1) {
    if (t < off) red[t] += red[t + off];
    __syncthreads();
  }
  float rs = rsqrtf(red[0] * (1.f / (float)E_) + 1e-5f);
  hb[(size_t)row * E_ + t]       = f2bf(d0 * rs * g[t] + bb[t]);
  hb[(size_t)row * E_ + t + 256] = f2bf(d1 * rs * g[t + 256] + bb[t + 256]);
  hb[(size_t)row * E_ + t + 512] = f2bf(d2 * rs * g[t + 512] + bb[t + 512]);
}

// ---- WMMA GEMM 3: m1 = gelu(hb(896x768) @ w1^T + b1) -> bf16 ---------------
__global__ void __launch_bounds__(256)
k_mlp1(const __bf16* hb, const __bf16* w1b, const float* b1, __bf16* m1b) {
  int wave = blockIdx.x * 8 + (threadIdx.x >> 5);
  int lane = threadIdx.x & 31;
  const int NP = NT_F / 2;                          // 96 strips of 32
  int m0 = (wave / NP) * 16, n0 = (wave % NP) * 32;
  v8f c0 = {0.f, 0.f, 0.f, 0.f, 0.f, 0.f, 0.f, 0.f};
  v8f c1 = c0;
  for (int k0 = 0; k0 < E_; k0 += 32) {
    v16bf a  = frag_a(hb, E_, m0, k0, lane);
    v16bf b0 = frag_bt(w1b, E_, n0, k0, lane);      // w1 is [F_][E_] == B^T
    v16bf b1v = frag_bt(w1b, E_, n0 + 16, k0, lane);
    c0 = wmma_bf16(a, b0, c0);
    c1 = wmma_bf16(a, b1v, c1);
  }
  int ln = lane & 15, half = lane >> 4;
#pragma unroll
  for (int j = 0; j < 2; ++j) {
    int col = n0 + j * 16 + ln;
    float bias = b1[col];
    v8f cc = j ? c1 : c0;
#pragma unroll
    for (int v = 0; v < 8; ++v) {
      int row = m0 + half * 8 + v;
      float xv = cc[v] + bias;
      float ge = 0.5f * xv * (1.f + erff(xv * 0.70710678118654752f)); // exact gelu
      m1b[(size_t)row * F_ + col] = f2bf(ge);
    }
  }
}

// ---- WMMA GEMM 4: tokens = tok0 + m1(896x3072) @ w2^T + b2 -> d_out --------
__global__ void __launch_bounds__(256)
k_mlp2(const __bf16* m1b, const __bf16* w2b, const float* b2, const float* tok0,
       float* out) {
  int wave = blockIdx.x * 8 + (threadIdx.x >> 5);
  int lane = threadIdx.x & 31;
  const int NP = NT_E / 2;                          // 24 strips of 32
  int m0 = (wave / NP) * 16, n0 = (wave % NP) * 32;
  v8f c0 = {0.f, 0.f, 0.f, 0.f, 0.f, 0.f, 0.f, 0.f};
  v8f c1 = c0;
  for (int k0 = 0; k0 < F_; k0 += 32) {
    v16bf a  = frag_a(m1b, F_, m0, k0, lane);
    v16bf b0 = frag_bt(w2b, F_, n0, k0, lane);      // w2 is [E_][F_] == B^T
    v16bf b1v = frag_bt(w2b, F_, n0 + 16, k0, lane);
    c0 = wmma_bf16(a, b0, c0);
    c1 = wmma_bf16(a, b1v, c1);
  }
  int ln = lane & 15, half = lane >> 4;
#pragma unroll
  for (int j = 0; j < 2; ++j) {
    int col = n0 + j * 16 + ln;
    float bias = b2[col];
    v8f cc = j ? c1 : c0;
#pragma unroll
    for (int v = 0; v < 8; ++v) {
      int row = m0 + half * 8 + v;
      out[(size_t)row * E_ + col] = cc[v] + bias + tok0[(size_t)row * E_ + col];
    }
  }
}

// ---------------------------------------------------------------------------
extern "C" void kernel_launch(void* const* d_in, const int* in_sizes, int n_in,
                              void* d_out, int out_size, void* d_ws, size_t ws_size,
                              hipStream_t stream) {
  const float* x       = (const float*)d_in[0];
  const float* cls     = (const float*)d_in[1];
  const float* wqkv_cp = (const float*)d_in[2];
  const float* bqkv_cp = (const float*)d_in[3];
  const float* wo_cp   = (const float*)d_in[4];
  const float* bo_cp   = (const float*)d_in[5];
  const float* wqkv_cc = (const float*)d_in[6];
  const float* bqkv_cc = (const float*)d_in[7];
  const float* wo_cc   = (const float*)d_in[8];
  const float* bo_cc   = (const float*)d_in[9];
  const float* ln_g    = (const float*)d_in[10];
  const float* ln_b    = (const float*)d_in[11];
  const float* w1      = (const float*)d_in[12];
  const float* b1      = (const float*)d_in[13];
  const float* w2      = (const float*)d_in[14];
  const float* b2      = (const float*)d_in[15];
  (void)in_sizes; (void)n_in; (void)out_size;

  float* out_tokens = (float*)d_out;                           // [64][14][768]
  float* out_attn   = (float*)d_out + (size_t)B_ * L_ * E_;    // [64][14][576]

  // workspace carve (~210 MB total, 256B aligned)
  char* p = (char*)d_ws;
  auto carve = [&](size_t bytes) {
    char* r = p; p += (bytes + 255) & ~(size_t)255; return (void*)r;
  };
  __bf16* xb      = (__bf16*)carve((size_t)B_ * E_ * S_ * 2);     // 56.6 MB
  __bf16* xbT     = (__bf16*)carve((size_t)B_ * S_ * E_ * 2);     // 56.6 MB
  __bf16* qkA     = (__bf16*)carve((size_t)MPAD * E_ * 2);
  __bf16* attnB   = (__bf16*)carve((size_t)B_ * MPAD * S_ * 2);   // 13.0 MB
  __bf16* w1b     = (__bf16*)carve((size_t)F_ * E_ * 2);
  __bf16* w2b     = (__bf16*)carve((size_t)E_ * F_ * 2);
  __bf16* hb      = (__bf16*)carve((size_t)TROWS * E_ * 2);
  __bf16* m1b     = (__bf16*)carve((size_t)TROWS * F_ * 2);
  float*  scoresP = (float*)carve((size_t)B_ * MPAD * S_ * 4);    // 26.0 MB
  float*  ctxP    = (float*)carve((size_t)B_ * MPAD * E_ * 4);    // 34.6 MB
  float*  q_cp    = (float*)carve((size_t)L_ * E_ * 4);
  float*  cc_qkv  = (float*)carve((size_t)L_ * 3 * E_ * 4);
  float*  cc_ctx  = (float*)carve((size_t)L_ * E_ * 4);
  float*  cc_out  = (float*)carve((size_t)L_ * E_ * 4);
  float*  ctxv    = (float*)carve((size_t)B_ * L_ * E_ * 4);
  float*  tok0    = (float*)carve((size_t)B_ * L_ * E_ * 4);
  (void)ws_size;

  // stage bf16 operands (both orientations of x)
  int nxb = B_ * E_ * S_;
  k_f2bf<<<nxb / 256, 256, 0, stream>>>(x, xb, nxb);
  k_transpose<<<B_ * (E_ / 64) * (S_ / 64), 256, 0, stream>>>(x, xbT);
  k_f2bf<<<(F_ * E_) / 256, 256, 0, stream>>>(w1, w1b, F_ * E_);
  k_f2bf<<<(E_ * F_) / 256, 256, 0, stream>>>(w2, w2b, E_ * F_);

  // batch-independent pieces
  k_rowdot<<<(L_ * E_) / 256, 256, 0, stream>>>(cls, wqkv_cp, bqkv_cp, q_cp,
                                                L_, E_, E_);
  k_rowdot<<<(L_ * 3 * E_) / 256, 256, 0, stream>>>(cls, wqkv_cc, bqkv_cc, cc_qkv,
                                                    L_, 3 * E_, E_);
  k_qkfold<<<(MPAD * E_) / 256, 256, 0, stream>>>(q_cp, wqkv_cp, qkA);
  k_ccattn<<<H_ * L_, 64, 0, stream>>>(cc_qkv, cc_ctx);
  k_rowdot<<<(L_ * E_) / 256, 256, 0, stream>>>(cc_ctx, wo_cc, bo_cc, cc_out,
                                                L_, E_, E_);

  // heavy path: scores -> softmax -> ctx (WMMA)
  k_scores<<<(B_ * MT_Q * (NT_S / 2)) / 8, 256, 0, stream>>>(qkA, xbT, scoresP);
  k_softmax<<<B_ * MPAD, 192, 0, stream>>>(scoresP, attnB);
  k_attnmean<<<(B_ * L_ * S_) / 256, 256, 0, stream>>>(scoresP, out_attn);
  k_ctx<<<(B_ * MT_Q * (NT_E / 2)) / 8, 256, 0, stream>>>(attnB, xb, ctxP);

  // head V-projection + output projection + residual
  k_vproj<<<(B_ * L_ * E_) / 256, 256, 0, stream>>>(ctxP, wqkv_cp, bqkv_cp, ctxv);
  k_oproj<<<(B_ * L_ * E_) / 256, 256, 0, stream>>>(ctxv, wo_cp, bo_cp, cls,
                                                    cc_out, tok0);

  // LN + MLP (WMMA) + residual -> tokens output
  k_layernorm<<<TROWS, 256, 0, stream>>>(tok0, ln_g, ln_b, hb);
  k_mlp1<<<(MT_T * (NT_F / 2)) / 8, 256, 0, stream>>>(hb, w1b, b1, m1b);
  k_mlp2<<<(MT_T * (NT_E / 2)) / 8, 256, 0, stream>>>(m1b, w2b, b2, tok0,
                                                      out_tokens);
}
